// LIFBoxBenchmark_68135361183762
// MI455X (gfx1250) — matrix operand order repaired and checked by reading, not compiled, not run on hardware
//
#include <hip/hip_runtime.h>

typedef __attribute__((ext_vector_type(16))) _Float16 v16h;
typedef __attribute__((ext_vector_type(8)))  float    v8f;

#define kT 256
#define kB 64
#define kF 1024

union AFrag { v16h v; _Float16 h[16]; };
union BFrag { v16h v; uint4 q[2]; };

// ---------------------------------------------------------------------------
// Kernel 1: W[g][f] fp32 (row-major) -> Wt[g][f] fp16 (K-major, the layout the
// WMMA B-matrix wants: lane for column N reads its K values from one row). Tiled via LDS so
// both the global read and the global write are coalesced.
// ---------------------------------------------------------------------------
__global__ __launch_bounds__(256) void snn_w_cvt_t(const float* __restrict__ W,
                                                   _Float16* __restrict__ Wt) {
  __shared__ float tile[32][33];
  const int tx = threadIdx.x;        // 0..31
  const int ty = threadIdx.y;        // 0..7
  const int kb = blockIdx.x * 32;    // k block
  const int nb = blockIdx.y * 32;    // n (=g) block
#pragma unroll
  for (int i = 0; i < 32; i += 8)
    tile[ty + i][tx] = W[(size_t)(nb + ty + i) * kF + kb + tx];
  __syncthreads();
#pragma unroll
  for (int i = 0; i < 32; i += 8)
    Wt[(size_t)(nb + ty + i) * kF + kb + tx] = (_Float16)tile[ty + i][tx];
}

__device__ __forceinline__ void cvt8(AFrag& a, int base, float4 lo, float4 hi) {
  a.h[base + 0] = (_Float16)lo.x;  a.h[base + 1] = (_Float16)lo.y;
  a.h[base + 2] = (_Float16)lo.z;  a.h[base + 3] = (_Float16)lo.w;
  a.h[base + 4] = (_Float16)hi.x;  a.h[base + 5] = (_Float16)hi.y;
  a.h[base + 6] = (_Float16)hi.z;  a.h[base + 7] = (_Float16)hi.w;
}

// ---------------------------------------------------------------------------
// Kernel 2: fused GEMM (WMMA f16 -> f32) + LIF membrane scan.
// Block = (g-tile of 64, batch b). Block computes cur[t, b, gtile] for ALL
// t = 0..255 (M dimension of the GEMM is exactly t, since b is fixed), parks
// it in a 64KB LDS tile, then 64 lanes run the sequential LIF recurrence down
// their column and emit spikes. This keeps the 64MB `cur` tensor out of HBM.
// 8 waves; wave w owns M rows [32w, 32w+32) x all 64 N columns = 2x4 tiles.
// ---------------------------------------------------------------------------
__global__ __launch_bounds__(256) void snn_lif_fused(const float* __restrict__ X,
                                                     const _Float16* __restrict__ Wt,
                                                     float* __restrict__ out) {
  __shared__ float s_cur[kT * 64];   // 64 KB: [T][64] tile of cur

  const int gbase = blockIdx.x * 64;
  const int b     = blockIdx.y;
  const int tid   = threadIdx.x;
  const int wave  = tid >> 5;
  const int lane  = tid & 31;
  const int lhalf = lane >> 4;   // which K-half this lane holds in the A frag
  const int lmod  = lane & 15;

  const int m0 = wave * 32;      // first T row owned by this wave

  v8f acc[2][4];
  const v8f vzero = {0.f, 0.f, 0.f, 0.f, 0.f, 0.f, 0.f, 0.f};
#pragma unroll
  for (int mt = 0; mt < 2; ++mt)
#pragma unroll
    for (int nt = 0; nt < 4; ++nt) acc[mt][nt] = vzero;

  // A rows: X[t, b, :] with t = m0 + lmod (+16 for the second M tile).
  // A-frag f16 layout (16x32): lane m holds K {0..7,16..23} (lanes 0-15) or
  // K {8..15,24..31} (lanes 16-31) -> per-lane fp32 loads at k + 8*lhalf.
  const float* xrow0 = X + ((size_t)(m0 + lmod) * kB + b) * kF + lhalf * 8;
  const float* xrow1 = xrow0 + (size_t)16 * kB * kF;
  // B-frag (32x16): lane holds column N = lmod, K {0..7,16..23} / {8..15,24..31} like A.
  const _Float16* wrow = Wt + (size_t)(gbase + lmod) * kF + lhalf * 8;

  for (int k0 = 0; k0 < kF; k0 += 32) {
    AFrag a[2];
    {
      const float* p = xrow0 + k0;
      float4 f0 = *(const float4*)(p);
      float4 f1 = *(const float4*)(p + 4);
      float4 f2 = *(const float4*)(p + 16);
      float4 f3 = *(const float4*)(p + 20);
      cvt8(a[0], 0, f0, f1);
      cvt8(a[0], 8, f2, f3);
      __builtin_prefetch((const void*)(p + 32), 0, 0);
    }
    {
      const float* p = xrow1 + k0;
      float4 f0 = *(const float4*)(p);
      float4 f1 = *(const float4*)(p + 4);
      float4 f2 = *(const float4*)(p + 16);
      float4 f3 = *(const float4*)(p + 20);
      cvt8(a[1], 0, f0, f1);
      cvt8(a[1], 8, f2, f3);
      __builtin_prefetch((const void*)(p + 32), 0, 0);
    }

    const _Float16* wk = wrow + k0;
    BFrag bf[4];
#pragma unroll
    for (int nt = 0; nt < 4; ++nt) {
      bf[nt].q[0] = *(const uint4*)(wk + (size_t)nt * 16 * kF);
      bf[nt].q[1] = *(const uint4*)(wk + (size_t)nt * 16 * kF + 16);
    }

#pragma unroll
    for (int mt = 0; mt < 2; ++mt)
#pragma unroll
      for (int nt = 0; nt < 4; ++nt)
        acc[mt][nt] = __builtin_amdgcn_wmma_f32_16x16x32_f16(
            false, a[mt].v, false, bf[nt].v, (short)0, acc[mt][nt],
            false, false);
  }

  // D layout (16x16 f32): VGPR j -> M = j + 8*(lane>=16), N = lane&15.
#pragma unroll
  for (int mt = 0; mt < 2; ++mt)
#pragma unroll
    for (int nt = 0; nt < 4; ++nt)
#pragma unroll
      for (int j = 0; j < 8; ++j) {
        const int m = m0 + mt * 16 + j + lhalf * 8;
        const int n = nt * 16 + lmod;
        s_cur[m * 64 + n] = acc[mt][nt][j];
      }
  __syncthreads();

  // LIF scan (tau=2, v_th=1, hard reset to 0): strictly sequential over t.
  // 64 lanes, one per g column; stores are coalesced 64x4B per step.
  if (tid < 64) {
    float v = 0.f;
    float* ob = out + (size_t)b * kF + gbase + tid;
    for (int t = 0; t < kT; ++t) {
      const float x = s_cur[t * 64 + tid];
      v = 0.5f * (v + x);                       // v += (x - v)/tau
      const float s = (v >= 1.0f) ? 1.0f : 0.0f; // heaviside(v - v_th)
      v = (1.0f - s) * v;                        // hard reset
      ob[(size_t)t * (kB * kF)] = s;
    }
  }
}

// ---------------------------------------------------------------------------
extern "C" void kernel_launch(void* const* d_in, const int* in_sizes, int n_in,
                              void* d_out, int out_size, void* d_ws, size_t ws_size,
                              hipStream_t stream) {
  const float* X  = (const float*)d_in[0];   // input_spikes [T,B,F] fp32
  const float* W  = (const float*)d_in[1];   // W [F,F] fp32
  float*       out = (float*)d_out;          // spikes [T,B,F] fp32
  _Float16*    Wt  = (_Float16*)d_ws;        // 2 MB: W^T in fp16, K-major

  snn_w_cvt_t<<<dim3(kF / 32, kF / 32), dim3(32, 8), 0, stream>>>(W, Wt);
  snn_lif_fused<<<dim3(kF / 64, kB), 256, 0, stream>>>(X, Wt, out);

  (void)in_sizes; (void)n_in; (void)out_size; (void)ws_size;
}
